// CloudCrop_4028679323865
// MI455X (gfx1250) — compile-verified, matrix operand order checked by reading, not run on hardware
//
#include <hip/hip_runtime.h>
#include <hip/hip_bf16.h>

// ---------------------------------------------------------------------------
// Problem constants (from reference): B=4, N=2048, C=256, OUT=256, S=32
// ---------------------------------------------------------------------------
#define BATCH   4
#define NPTS    2048
#define CIN     256
#define COUT    256
#define NSAMP   32
#define K1RAW   259            // 3 + CIN
#define K1PAD   288            // padded to multiple of 32 for bf16 WMMA
#define K2PAD   256
#define NCOL    262144L        // BATCH*NPTS*NSAMP
#define LDSROW  40             // ushorts per LDS tile row (80 B; 32 data + 8 pad)
#define RADIUS_F  0.05f
#define HMIN_F   (-0.02f)
#define HMAX_F    0.04f
#define EPS_F     1e-5f

typedef __attribute__((ext_vector_type(16))) __bf16 v16bf;
typedef __attribute__((ext_vector_type(8)))  float  v8f;

struct alignas(16) U4 { unsigned int d[4]; };
union Frag { v16bf v; U4 q[2]; };

static __device__ __forceinline__ unsigned short f32_to_bf16(float f) {
    unsigned int u = __float_as_uint(f);
    u += 0x7FFFu + ((u >> 16) & 1u);          // round-to-nearest-even
    return (unsigned short)(u >> 16);
}

// CDNA5 async global->LDS copy (ASYNCcnt-tracked). VDST = LDS byte offset,
// VADDR = 64-bit global address (GV mode). 16 bytes per lane.
static __device__ __forceinline__ void async_b128(unsigned lds_off, const void* gptr) {
    asm volatile("global_load_async_to_lds_b128 %0, %1, off"
                 :: "v"(lds_off), "v"(gptr) : "memory");
}
static __device__ __forceinline__ void wait_async0() {
    asm volatile("s_wait_asynccnt 0x0" ::: "memory");
}

// ---------------------------------------------------------------------------
// 1) Cylinder query: one wave32 per query point (b,p). Ordered first-32
//    selection via ballot/popc ranks == reference top_k(-keys) semantics.
// ---------------------------------------------------------------------------
__global__ __launch_bounds__(256)
void cylinder_query_kernel(const float* __restrict__ xyz,
                           const float* __restrict__ rot,
                           int* __restrict__ idx) {
    const int lane = threadIdx.x & 31;
    const int wave = threadIdx.x >> 5;
    const int gid  = blockIdx.x * 8 + wave;      // b*NPTS + p
    const int b    = gid >> 11;

    const float* q = xyz + (size_t)gid * 3;
    const float qx = q[0], qy = q[1], qz = q[2];
    const float* R = rot + (size_t)gid * 9;
    const float r00 = R[0], r01 = R[1], r02 = R[2];
    const float r10 = R[3], r11 = R[4], r12 = R[5];
    const float r20 = R[6], r21 = R[7], r22 = R[8];
    const float* bx = xyz + (size_t)b * NPTS * 3;

    int* out  = idx + (size_t)gid * NSAMP;
    int count = 0, first = -1;

    for (int c = 0; c < NPTS; c += 32) {
        const int j = c + lane;
        const float dx = bx[j * 3 + 0] - qx;
        const float dy = bx[j * 3 + 1] - qy;
        const float dz = bx[j * 3 + 2] - qz;
        const float a0 = r00 * dx + r01 * dy + r02 * dz;
        const float a1 = r10 * dx + r11 * dy + r12 * dz;
        const float a2 = r20 * dx + r21 * dy + r22 * dz;
        const bool m = (a1 * a1 + a2 * a2 < RADIUS_F * RADIUS_F) &&
                       (a0 > HMIN_F) && (a0 < HMAX_F);
        const unsigned int bal = (unsigned int)__ballot(m);
        if (first < 0 && bal) first = c + __ffs(bal) - 1;
        const int rank = __popc(bal & ((1u << lane) - 1u));
        const int slot = count + rank;
        if (m && slot < NSAMP) out[slot] = j;
        count += __popc(bal);
        if (count >= NSAMP) break;
    }
    const int nv   = count < NSAMP ? count : NSAMP;
    const int fill = (count > 0) ? first : 0;
    if (lane >= nv) out[lane] = fill;
}

// ---------------------------------------------------------------------------
// 2) Group + pack to bf16: G[col][K1PAD], col = (b*NPTS+p)*NSAMP + s.
// ---------------------------------------------------------------------------
__global__ __launch_bounds__(256)
void group_pack_kernel(const float* __restrict__ xyz,
                       const float* __restrict__ feat,
                       const float* __restrict__ rot,
                       const int* __restrict__ idx,
                       unsigned short* __restrict__ G) {
    const int gid  = blockIdx.x;                 // b*NPTS + p
    const int b    = gid >> 11;
    const int t    = threadIdx.x;
    const int s    = t >> 3;                     // 0..31
    const int part = t & 7;                      // 0..7
    const int j    = idx[(size_t)gid * NSAMP + s];
    const size_t col = (size_t)gid * NSAMP + s;
    unsigned short* gcol = G + col * K1PAD;

    const float* f = feat + (size_t)b * CIN * NPTS;
    const int c0 = part * 32;
    #pragma unroll 8
    for (int c = 0; c < 32; ++c)
        gcol[3 + c0 + c] = f32_to_bf16(f[(size_t)(c0 + c) * NPTS + j]);

    if (part == 0) {
        const float* q  = xyz + (size_t)gid * 3;
        const float* pj = xyz + ((size_t)b * NPTS + j) * 3;
        const float inv = 1.0f / RADIUS_F;
        const float dx = (pj[0] - q[0]) * inv;
        const float dy = (pj[1] - q[1]) * inv;
        const float dz = (pj[2] - q[2]) * inv;
        const float* R = rot + (size_t)gid * 9;   // row-major [j][k]
        gcol[0] = f32_to_bf16(dx * R[0] + dy * R[3] + dz * R[6]);
        gcol[1] = f32_to_bf16(dx * R[1] + dy * R[4] + dz * R[7]);
        gcol[2] = f32_to_bf16(dx * R[2] + dy * R[5] + dz * R[8]);
        for (int c = K1RAW; c < K1PAD; ++c) gcol[c] = 0;
    }
}

// ---------------------------------------------------------------------------
// 3) Weight conversion f32 -> bf16 (with K padding for W1)
// ---------------------------------------------------------------------------
__global__ void pack_weights_kernel(const float* __restrict__ W1,
                                    const float* __restrict__ W2,
                                    unsigned short* __restrict__ W1b,
                                    unsigned short* __restrict__ W2b) {
    const int o = blockIdx.x;        // 0..255
    const int t = threadIdx.x;       // 0..287
    W1b[(size_t)o * K1PAD + t] =
        (t < K1RAW) ? f32_to_bf16(W1[(size_t)o * K1RAW + t]) : (unsigned short)0;
    if (t < K2PAD)
        W2b[(size_t)o * K2PAD + t] = f32_to_bf16(W2[(size_t)o * K2PAD + t]);
}

// ---------------------------------------------------------------------------
// 4) bf16 WMMA GEMM with double-buffered async global->LDS tile staging.
//    Y[256][NCOL] = A[256][Kpad] * B[NCOL][Kpad]^T
//    Block = 8 waves in 2(M) x 4(N); wave tile 64x32 = 8 acc tiles;
//    8 x v_wmma_f32_16x16x32_bf16 per K=32 step.
//    Tiles: A 128x32 bf16, B 128x32 bf16 in LDS, row stride 80 B (pad ->
//    16-lane conflict-free ds_load_b128). Copies are 64 B runs per 4 lanes
//    (coalesced), issued with global_load_async_to_lds_b128 and fenced with
//    s_wait_asynccnt + workgroup barrier; next tile overlaps current WMMAs.
// ---------------------------------------------------------------------------
__global__ __launch_bounds__(256)
void wmma_gemm_kernel(const unsigned short* __restrict__ A,
                      const unsigned short* __restrict__ Bm,
                      float* __restrict__ Y,
                      int Kpad) {
    __shared__ unsigned short sA[2][128 * LDSROW];
    __shared__ unsigned short sB[2][128 * LDSROW];

    const int t    = threadIdx.x;
    const int lane = t & 31;
    const int wave = t >> 5;
    const int wm = wave >> 2, wn = wave & 3;
    const int l15 = lane & 15;
    const int kh  = (lane < 16) ? 0 : 8;        // K-half split per ISA layout
    const int nBase = blockIdx.x * 128;
    const int mBase = blockIdx.y * 128;

    // Cooperative copy mapping: 512 16B-chunks per tile; thread t takes
    // chunk t (row t>>2, part t&3) and chunk t+256 (row +64).
    const int r0 = t >> 2;
    const int p0 = t & 3;

    const unsigned short* gA0 = A  + (size_t)(mBase + r0) * Kpad + p0 * 8;
    const unsigned short* gA1 = gA0 + (size_t)64 * Kpad;
    const unsigned short* gB0 = Bm + (size_t)(nBase + r0) * Kpad + p0 * 8;
    const unsigned short* gB1 = gB0 + (size_t)64 * Kpad;

    unsigned lA0[2], lA1[2], lB0[2], lB1[2];
    #pragma unroll
    for (int bi = 0; bi < 2; ++bi) {
        lA0[bi] = (unsigned)(uintptr_t)&sA[bi][(size_t)r0 * LDSROW + p0 * 8];
        lA1[bi] = (unsigned)(uintptr_t)&sA[bi][(size_t)(r0 + 64) * LDSROW + p0 * 8];
        lB0[bi] = (unsigned)(uintptr_t)&sB[bi][(size_t)r0 * LDSROW + p0 * 8];
        lB1[bi] = (unsigned)(uintptr_t)&sB[bi][(size_t)(r0 + 64) * LDSROW + p0 * 8];
    }

    auto issue = [&](int bi, int kb) {
        async_b128(lA0[bi], gA0 + kb);
        async_b128(lA1[bi], gA1 + kb);
        async_b128(lB0[bi], gB0 + kb);
        async_b128(lB1[bi], gB1 + kb);
    };

    const v8f vzero = {0.f, 0.f, 0.f, 0.f, 0.f, 0.f, 0.f, 0.f};
    v8f acc[4][2];
    #pragma unroll
    for (int i = 0; i < 4; ++i)
        #pragma unroll
        for (int j = 0; j < 2; ++j) acc[i][j] = vzero;

    issue(0, 0);
    int buf = 0;
    for (int kb = 0; kb < Kpad; kb += 32) {
        wait_async0();          // our copies into 'buf' have landed
        __syncthreads();        // everyone's copies landed; prev reads done
        if (kb + 32 < Kpad) issue(buf ^ 1, kb + 32);

        Frag af[4], bfr[2];
        #pragma unroll
        for (int fm = 0; fm < 4; ++fm) {
            const unsigned short* p =
                &sA[buf][(size_t)(wm * 64 + fm * 16 + l15) * LDSROW + kh];
            af[fm].q[0] = *(const U4*)p;
            af[fm].q[1] = *(const U4*)(p + 16);
        }
        #pragma unroll
        for (int fn = 0; fn < 2; ++fn) {
            const unsigned short* p =
                &sB[buf][(size_t)(wn * 32 + fn * 16 + l15) * LDSROW + kh];
            bfr[fn].q[0] = *(const U4*)p;
            bfr[fn].q[1] = *(const U4*)(p + 16);
        }
        #pragma unroll
        for (int fm = 0; fm < 4; ++fm)
            #pragma unroll
            for (int fn = 0; fn < 2; ++fn)
                acc[fm][fn] = __builtin_amdgcn_wmma_f32_16x16x32_bf16(
                    false, af[fm].v, false, bfr[fn].v,
                    (short)0, acc[fm][fn], false, false);
        buf ^= 1;
    }

    const int rowOff = (lane < 16) ? 0 : 8;
    #pragma unroll
    for (int fm = 0; fm < 4; ++fm)
        #pragma unroll
        for (int fn = 0; fn < 2; ++fn)
            #pragma unroll
            for (int v = 0; v < 8; ++v) {
                const int row = mBase + wm * 64 + fm * 16 + v + rowOff;
                const size_t col = (size_t)nBase + wn * 32 + fn * 16 + l15;
                Y[(size_t)row * NCOL + col] = acc[fm][fn][v];
            }
}

// ---------------------------------------------------------------------------
// 5) Per-channel sum / sum-of-squares (BatchNorm statistics)
// ---------------------------------------------------------------------------
__global__ __launch_bounds__(256)
void stats_kernel(const float* __restrict__ Y,
                  float* __restrict__ sum, float* __restrict__ sumsq) {
    const int o = blockIdx.x;
    const float* row = Y + (size_t)o * NCOL;
    float s = 0.f, ss = 0.f;
    const long t = threadIdx.x + (long)blockIdx.y * 256;
    for (long i = t; i < NCOL; i += 256L * 32L) {
        const float v = row[i];
        s += v; ss += v * v;
    }
    __shared__ float sb[256], qb[256];
    sb[threadIdx.x] = s; qb[threadIdx.x] = ss;
    __syncthreads();
    for (int off = 128; off > 0; off >>= 1) {
        if ((int)threadIdx.x < off) {
            sb[threadIdx.x] += sb[threadIdx.x + off];
            qb[threadIdx.x] += qb[threadIdx.x + off];
        }
        __syncthreads();
    }
    if (threadIdx.x == 0) {
        atomicAdd(&sum[o], sb[0]);
        atomicAdd(&sumsq[o], qb[0]);
    }
}

__global__ void scaleshift_kernel(const float* __restrict__ sum,
                                  const float* __restrict__ sumsq,
                                  const float* __restrict__ gamma,
                                  const float* __restrict__ beta,
                                  float* __restrict__ scale,
                                  float* __restrict__ shift) {
    const int o = threadIdx.x;
    const float M = (float)NCOL;
    const float mu  = sum[o] / M;
    const float var = sumsq[o] / M - mu * mu;
    const float sc  = gamma[o] * rsqrtf(var + EPS_F);
    scale[o] = sc;
    shift[o] = beta[o] - mu * sc;
}

// ---------------------------------------------------------------------------
// 6) BN + ReLU + repack to bf16 column-K layout for GEMM2 input
// ---------------------------------------------------------------------------
__global__ __launch_bounds__(256)
void bn_relu_pack_kernel(const float* __restrict__ Y,
                         const float* __restrict__ scale,
                         const float* __restrict__ shift,
                         unsigned short* __restrict__ H) {
    const size_t col = (size_t)blockIdx.x * 256 + threadIdx.x;
    unsigned short* h = H + col * K2PAD;
    for (int o = 0; o < COUT; ++o) {
        float v = Y[(size_t)o * NCOL + col] * scale[o] + shift[o];
        v = v > 0.f ? v : 0.f;
        h[o] = f32_to_bf16(v);
    }
}

// ---------------------------------------------------------------------------
// 7) BN + ReLU + max over S -> out (B, COUT, N) f32
// ---------------------------------------------------------------------------
__global__ __launch_bounds__(256)
void bn_relu_max_kernel(const float* __restrict__ Y,
                        const float* __restrict__ scale,
                        const float* __restrict__ shift,
                        float* __restrict__ out) {
    const long tid = (long)blockIdx.x * 256 + threadIdx.x; // (b,o,p)
    const int p = (int)(tid & (NPTS - 1));
    const int o = (int)((tid >> 11) & (COUT - 1));
    const int b = (int)(tid >> 19);
    const float sc = scale[o], sh = shift[o];
    const float* y = Y + (size_t)o * NCOL + ((size_t)(b * NPTS + p)) * NSAMP;
    float m = 0.f;                                // relu floor
    #pragma unroll 8
    for (int s = 0; s < NSAMP; ++s) {
        const float v = y[s] * sc + sh;
        if (v > m) m = v;
    }
    out[(size_t)b * COUT * NPTS + (size_t)o * NPTS + p] = m;
}

// ---------------------------------------------------------------------------
// Host-side launcher
// ---------------------------------------------------------------------------
extern "C" void kernel_launch(void* const* d_in, const int* in_sizes, int n_in,
                              void* d_out, int out_size, void* d_ws, size_t ws_size,
                              hipStream_t stream) {
    const float* xyz    = (const float*)d_in[0];
    const float* feat   = (const float*)d_in[1];
    const float* rot    = (const float*)d_in[2];
    const float* W1     = (const float*)d_in[3];
    const float* gamma1 = (const float*)d_in[4];
    const float* beta1  = (const float*)d_in[5];
    const float* W2     = (const float*)d_in[6];
    const float* gamma2 = (const float*)d_in[7];
    const float* beta2  = (const float*)d_in[8];
    float* out = (float*)d_out;

    char* ws = (char*)d_ws;
    size_t off = 0;
    auto carve = [&](size_t bytes) -> char* {
        char* p = ws + off;
        off = (off + bytes + 255) & ~(size_t)255;
        return p;
    };
    int*            idx  = (int*)           carve((size_t)BATCH * NPTS * NSAMP * 4);
    unsigned short* W1b  = (unsigned short*)carve((size_t)COUT * K1PAD * 2);
    unsigned short* W2b  = (unsigned short*)carve((size_t)COUT * K2PAD * 2);
    float*          sum1 = (float*)carve(COUT * 4);
    float*          sq1  = (float*)carve(COUT * 4);
    float*          sum2 = (float*)carve(COUT * 4);
    float*          sq2  = (float*)carve(COUT * 4);
    float*          sc1  = (float*)carve(COUT * 4);
    float*          sh1  = (float*)carve(COUT * 4);
    float*          sc2  = (float*)carve(COUT * 4);
    float*          sh2  = (float*)carve(COUT * 4);
    unsigned short* G    = (unsigned short*)carve((size_t)NCOL * K1PAD * 2);
    unsigned short* H    = (unsigned short*)carve((size_t)NCOL * K2PAD * 2);
    float*          Y    = (float*)carve((size_t)COUT * NCOL * 4); // reused for both layers

    // zero the atomic-accumulated stats (deterministic across replays)
    hipMemsetAsync(sum1, 0, COUT * 4, stream);
    hipMemsetAsync(sq1,  0, COUT * 4, stream);
    hipMemsetAsync(sum2, 0, COUT * 4, stream);
    hipMemsetAsync(sq2,  0, COUT * 4, stream);

    pack_weights_kernel<<<COUT, K1PAD, 0, stream>>>(W1, W2, W1b, W2b);

    cylinder_query_kernel<<<(BATCH * NPTS) / 8, 256, 0, stream>>>(xyz, rot, idx);

    group_pack_kernel<<<BATCH * NPTS, 256, 0, stream>>>(xyz, feat, rot, idx, G);

    // Layer 1
    wmma_gemm_kernel<<<dim3(NCOL / 128, COUT / 128), 256, 0, stream>>>(W1b, G, Y, K1PAD);
    stats_kernel<<<dim3(COUT, 32), 256, 0, stream>>>(Y, sum1, sq1);
    scaleshift_kernel<<<1, COUT, 0, stream>>>(sum1, sq1, gamma1, beta1, sc1, sh1);
    bn_relu_pack_kernel<<<NCOL / 256, 256, 0, stream>>>(Y, sc1, sh1, H);

    // Layer 2 (Y reused)
    wmma_gemm_kernel<<<dim3(NCOL / 128, COUT / 128), 256, 0, stream>>>(W2b, H, Y, K2PAD);
    stats_kernel<<<dim3(COUT, 32), 256, 0, stream>>>(Y, sum2, sq2);
    scaleshift_kernel<<<1, COUT, 0, stream>>>(sum2, sq2, gamma2, beta2, sc2, sh2);

    bn_relu_max_kernel<<<(BATCH * COUT * NPTS) / 256, 256, 0, stream>>>(Y, sc2, sh2, out);
}
